// Attention_9174050144962
// MI455X (gfx1250) — compile-verified, matrix operand order checked by reading, not run on hardware
//
#include <hip/hip_runtime.h>
#include <hip/hip_bf16.h>

// ---------------------------------------------------------------------------
// Problem constants (from reference)
// ---------------------------------------------------------------------------
#define B_SZ     128
#define T_SZ     26
#define IN_SZ    512
#define H_SZ     512
#define NC_SZ    38
#define STEPS_SZ 26
#define BPE_SZ   50257
#define WP_SZ    30522
#define G4H      2048          // 4*H
#define KIN      550           // IN + NC
#define KCAT     1088          // 512 + 38 + 512 = 1062, padded to mult of 32
#define MROWS    (B_SZ * T_SZ) // 3328

typedef __bf16 bf16_t;
typedef __attribute__((ext_vector_type(16))) __bf16 v16bf;
typedef __attribute__((ext_vector_type(8)))  float  v8f;

// ---------------------------------------------------------------------------
// WMMA bf16 fragment loaders (CDNA5 16-bit A 16x32 layout; B mirrored by col)
// Per lane: half = lane>>4 ; elements 0..7  = K (kb + half*8 + 0..7)
//                            elements 8..15 = K (kb + half*8 + 16..23)
// Two contiguous 16-byte groups -> global_load_b128 pairs.
// ---------------------------------------------------------------------------
__device__ __forceinline__ v16bf frag_bf16(const bf16_t* row, int kb, int half) {
  const bf16_t* q = row + kb + half * 8;
  union { uint4 u[2]; v16bf v; } t;
  t.u[0] = *(const uint4*)(q);
  t.u[1] = *(const uint4*)(q + 16);
  return t.v;
}

__device__ __forceinline__ v16bf frag_f32(const float* row, int kb, int half) {
  const float* q = row + kb + half * 8;
  float4 a0 = *(const float4*)(q);
  float4 a1 = *(const float4*)(q + 4);
  float4 b0 = *(const float4*)(q + 16);
  float4 b1 = *(const float4*)(q + 20);
  v16bf v;
  v[0]  = (bf16_t)a0.x; v[1]  = (bf16_t)a0.y; v[2]  = (bf16_t)a0.z; v[3]  = (bf16_t)a0.w;
  v[4]  = (bf16_t)a1.x; v[5]  = (bf16_t)a1.y; v[6]  = (bf16_t)a1.z; v[7]  = (bf16_t)a1.w;
  v[8]  = (bf16_t)b0.x; v[9]  = (bf16_t)b0.y; v[10] = (bf16_t)b0.z; v[11] = (bf16_t)b0.w;
  v[12] = (bf16_t)b1.x; v[13] = (bf16_t)b1.y; v[14] = (bf16_t)b1.z; v[15] = (bf16_t)b1.w;
  return v;
}

// ---------------------------------------------------------------------------
// Generic WMMA GEMM:  O[M,N] = A[M,K] * B[N,K]^T (+ bias[n])
// Block: 256 threads = 8 waves arranged 2(M) x 4(N).
// Wave tile: (MT*16) x 64.  Block tile: (MT*32) x 256.
// MT=2 for the big GEMMs (B-frag amortized over 2 WMMAs), MT=1 for the small
// latency-bound recurrent GEMM (more blocks in flight).
// M must be a multiple of MT*32 (3328 = 52*64 = 104*32; 128 = 4*32); N guarded.
// ---------------------------------------------------------------------------
template <bool AF32, bool BF32, bool BIAS, int MT>
__global__ void wmma_gemm(const void* __restrict__ Ap,
                          const void* __restrict__ Bp,
                          const float* __restrict__ bias,
                          float* __restrict__ O,
                          int M, int N, int K, int lda, int ldb, int ldo) {
  const int lane = threadIdx.x & 31;
  const int wave = threadIdx.x >> 5;
  const int half = lane >> 4;
  const int m0 = (blockIdx.y * 2 + (wave >> 2)) * (MT * 16);
  const int n0 = blockIdx.x * 256 + (wave & 3) * 64;
  if (m0 >= M) return;

  int mrow[MT];
#pragma unroll
  for (int mi = 0; mi < MT; ++mi) mrow[mi] = m0 + mi * 16 + (lane & 15);

  int nrow[4];
#pragma unroll
  for (int j = 0; j < 4; ++j) {
    int n = n0 + j * 16 + (lane & 15);
    nrow[j] = n < N ? n : (N - 1);
  }

  v8f acc[MT][4];
#pragma unroll
  for (int mi = 0; mi < MT; ++mi)
#pragma unroll
    for (int j = 0; j < 4; ++j) acc[mi][j] = (v8f){};

  for (int kb = 0; kb < K; kb += 32) {
    v16bf a[MT];
#pragma unroll
    for (int mi = 0; mi < MT; ++mi) {
      if constexpr (AF32) a[mi] = frag_f32((const float*)Ap + (size_t)mrow[mi] * lda, kb, half);
      else                a[mi] = frag_bf16((const bf16_t*)Ap + (size_t)mrow[mi] * lda, kb, half);
    }

#pragma unroll
    for (int j = 0; j < 4; ++j) {
      v16bf b;
      if constexpr (BF32) {
        const float* brow = (const float*)Bp + (size_t)nrow[j] * ldb;
        b = frag_f32(brow, kb, half);
        if (kb + 32 < K) __builtin_prefetch(brow + kb + 32, 0, 1);
      } else {
        b = frag_bf16((const bf16_t*)Bp + (size_t)nrow[j] * ldb, kb, half);
      }
#pragma unroll
      for (int mi = 0; mi < MT; ++mi)
        acc[mi][j] = __builtin_amdgcn_wmma_f32_16x16x32_bf16(
            false, a[mi], false, b, (short)0, acc[mi][j], false, false);
    }
  }

#pragma unroll
  for (int j = 0; j < 4; ++j) {
    int n = n0 + j * 16 + (lane & 15);
    if (n >= N) continue;
    float bv = BIAS ? bias[n] : 0.0f;
#pragma unroll
    for (int mi = 0; mi < MT; ++mi) {
#pragma unroll
      for (int r = 0; r < 8; ++r) {
        int m = m0 + mi * 16 + r + half * 8;
        O[(size_t)m * ldo + n] = acc[mi][j][r] + bv;
      }
    }
  }
}

// ---------------------------------------------------------------------------
// Prep: combined LSTM weight [Wih | Whh] -> bf16 [2048 x 1088], combined bias,
// zero h/c state, zero the x_cat K-pad columns (written once per launch).
// ---------------------------------------------------------------------------
__global__ void prep_kernel(const float* __restrict__ Wih,
                            const float* __restrict__ bih,
                            const float* __restrict__ Whh,
                            const float* __restrict__ bhh,
                            bf16_t* __restrict__ Wg, float* __restrict__ bg,
                            float* __restrict__ h, float* __restrict__ c,
                            bf16_t* __restrict__ xcat) {
  size_t idx = (size_t)blockIdx.x * 256 + threadIdx.x;
  if (idx < (size_t)G4H * KCAT) {
    int row = (int)(idx / KCAT);
    int col = (int)(idx % KCAT);
    float v = 0.0f;
    if (col < KIN)             v = Wih[(size_t)row * KIN + col];
    else if (col < KIN + H_SZ) v = Whh[(size_t)row * H_SZ + (col - KIN)];
    Wg[idx] = (bf16_t)v;
  }
  if (idx < G4H) bg[idx] = bih[idx] + bhh[idx];
  if (idx < (size_t)B_SZ * H_SZ) { h[idx] = 0.0f; c[idx] = 0.0f; }
  if (idx < (size_t)B_SZ * (KCAT - (KIN + H_SZ))) {
    int row = (int)(idx / (KCAT - (KIN + H_SZ)));
    int cc  = (int)(idx % (KCAT - (KIN + H_SZ)));
    xcat[(size_t)row * KCAT + (KIN + H_SZ) + cc] = (bf16_t)0.0f;
  }
}

// ---------------------------------------------------------------------------
// Attention step (one block per batch row, 256 threads):
//   ph = h @ h2h^T + b ; e_t = score . tanh(Hproj[b,t]+ph) ; softmax ;
//   context = sum_t alpha_t * batch_H[b,t] ; write bf16 x_cat row.
// ---------------------------------------------------------------------------
__global__ void attention_step(const float* __restrict__ batch_H,
                               const float* __restrict__ Hproj,
                               const float* __restrict__ h2h_W,
                               const float* __restrict__ h2h_b,
                               const float* __restrict__ score_W,
                               const int* __restrict__ text,
                               const float* __restrict__ h,
                               bf16_t* __restrict__ xcat, int t) {
  __shared__ float hsh[H_SZ];
  __shared__ float ph[H_SZ];
  __shared__ float sc[T_SZ];
  const int b = blockIdx.x;
  const int tid = threadIdx.x;
  const int lane = tid & 31;
  const int wave = tid >> 5;
  const float* hr = h + (size_t)b * H_SZ;

  for (int j = tid; j < H_SZ; j += 256) hsh[j] = hr[j];
  __syncthreads();

  for (int j = tid; j < H_SZ; j += 256) {
    const float* wr = h2h_W + (size_t)j * H_SZ;
    float s = h2h_b[j];
#pragma unroll 2
    for (int k = 0; k < H_SZ; k += 4) {
      float4 w4 = *(const float4*)(wr + k);
      s += hsh[k] * w4.x + hsh[k + 1] * w4.y + hsh[k + 2] * w4.z + hsh[k + 3] * w4.w;
    }
    ph[j] = s;
  }
  __syncthreads();

  for (int tt = wave; tt < T_SZ; tt += 8) {
    const float* hp = Hproj + ((size_t)b * T_SZ + tt) * H_SZ;
    float s = 0.0f;
    for (int k = lane; k < H_SZ; k += 32) s += tanhf(hp[k] + ph[k]) * score_W[k];
#pragma unroll
    for (int off = 16; off > 0; off >>= 1) s += __shfl_down(s, off, 32);
    if (lane == 0) sc[tt] = s;
  }
  __syncthreads();

  if (tid < 32) {
    float m = -1e30f;
    if (lane < T_SZ) m = sc[lane];
#pragma unroll
    for (int off = 16; off > 0; off >>= 1) m = fmaxf(m, __shfl_down(m, off, 32));
    m = __shfl(m, 0, 32);
    float e = (lane < T_SZ) ? __expf(sc[lane] - m) : 0.0f;
    float ss = e;
#pragma unroll
    for (int off = 16; off > 0; off >>= 1) ss += __shfl_down(ss, off, 32);
    ss = __shfl(ss, 0, 32);
    if (lane < T_SZ) sc[lane] = e / ss;
  }
  __syncthreads();

  bf16_t* xr = xcat + (size_t)b * KCAT;
  for (int j = tid; j < H_SZ; j += 256) {
    const float* bh = batch_H + (size_t)b * T_SZ * H_SZ + j;
    float cx = 0.0f;
#pragma unroll
    for (int tt = 0; tt < T_SZ; ++tt) cx += sc[tt] * bh[(size_t)tt * H_SZ];
    xr[j] = (bf16_t)cx;                     // context -> cols [0,512)
    xr[KIN + j] = (bf16_t)hsh[j];           // h       -> cols [550,1062)
  }
  int cls = text[b * STEPS_SZ + t];
  for (int j = tid; j < NC_SZ; j += 256)    // onehot  -> cols [512,550)
    xr[IN_SZ + j] = (bf16_t)((j == cls) ? 1.0f : 0.0f);
}

// ---------------------------------------------------------------------------
// LSTM pointwise: gate nonlinearity, state update, bf16 hidden for heads.
// ---------------------------------------------------------------------------
__device__ __forceinline__ float sigm(float x) { return 1.0f / (1.0f + __expf(-x)); }

__global__ void lstm_pointwise(const float* __restrict__ gates,
                               float* __restrict__ h, float* __restrict__ c,
                               bf16_t* __restrict__ hs, int t) {
  int idx = blockIdx.x * 256 + threadIdx.x;   // 0 .. 128*512-1
  int b = idx >> 9;
  int j = idx & 511;
  const float* g = gates + (size_t)b * G4H;
  float ig = g[j], fg = g[512 + j], gg = g[1024 + j], og = g[1536 + j];
  float cn = sigm(fg) * c[idx] + sigm(ig) * tanhf(gg);
  float hn = sigm(og) * tanhf(cn);
  c[idx] = cn;
  h[idx] = hn;
  hs[((size_t)b * T_SZ + t) * H_SZ + j] = (bf16_t)hn;
}

// ---------------------------------------------------------------------------
// Host: launch sequence
// ---------------------------------------------------------------------------
extern "C" void kernel_launch(void* const* d_in, const int* in_sizes, int n_in,
                              void* d_out, int out_size, void* d_ws, size_t ws_size,
                              hipStream_t stream) {
  const float* batch_H = (const float*)d_in[0];
  const int*   text    = (const int*)d_in[1];
  const float* i2h_W   = (const float*)d_in[2];
  const float* h2h_W   = (const float*)d_in[3];
  const float* h2h_b   = (const float*)d_in[4];
  const float* score_W = (const float*)d_in[5];
  const float* rnn_Wih = (const float*)d_in[6];
  const float* rnn_bih = (const float*)d_in[7];
  const float* rnn_Whh = (const float*)d_in[8];
  const float* rnn_bhh = (const float*)d_in[9];
  const float* char_W  = (const float*)d_in[10];
  const float* char_b  = (const float*)d_in[11];
  const float* bpe_W   = (const float*)d_in[12];
  const float* bpe_b   = (const float*)d_in[13];
  const float* wp_W    = (const float*)d_in[14];
  const float* wp_b    = (const float*)d_in[15];

  // Workspace layout (256B aligned regions), ~16.5 MB total
  auto align256 = [](size_t x) { return (x + 255) & ~(size_t)255; };
  char* ws = (char*)d_ws;
  size_t off = 0;
  float*  Hproj = (float*)(ws + off);  off += align256((size_t)MROWS * H_SZ * 4);
  bf16_t* xcat  = (bf16_t*)(ws + off); off += align256((size_t)B_SZ * KCAT * 2);
  bf16_t* Wg    = (bf16_t*)(ws + off); off += align256((size_t)G4H * KCAT * 2);
  float*  bg    = (float*)(ws + off);  off += align256((size_t)G4H * 4);
  float*  gates = (float*)(ws + off);  off += align256((size_t)B_SZ * G4H * 4);
  float*  hbuf  = (float*)(ws + off);  off += align256((size_t)B_SZ * H_SZ * 4);
  float*  cbuf  = (float*)(ws + off);  off += align256((size_t)B_SZ * H_SZ * 4);
  bf16_t* hsb   = (bf16_t*)(ws + off); off += align256((size_t)MROWS * H_SZ * 2);
  (void)ws_size; (void)n_in; (void)in_sizes; (void)out_size;

  // 1) Prep: bf16 combined LSTM weights, combined bias, zero state & pad
  {
    size_t total = (size_t)G4H * KCAT;
    int blocks = (int)((total + 255) / 256);
    prep_kernel<<<blocks, 256, 0, stream>>>(rnn_Wih, rnn_bih, rnn_Whh, rnn_bhh,
                                            Wg, bg, hbuf, cbuf, xcat);
  }

  // 2) Hproj[3328,512] = batch_H @ i2h_W^T  (f32 -> bf16 on the fly)
  wmma_gemm<true, true, false, 2><<<dim3(H_SZ / 256, MROWS / 64), 256, 0, stream>>>(
      batch_H, i2h_W, nullptr, Hproj, MROWS, H_SZ, IN_SZ, IN_SZ, IN_SZ, H_SZ);

  // 3) Recurrent scan: 26 sequential steps
  for (int t = 0; t < STEPS_SZ; ++t) {
    attention_step<<<B_SZ, 256, 0, stream>>>(batch_H, Hproj, h2h_W, h2h_b,
                                             score_W, text, hbuf, xcat, t);
    wmma_gemm<false, false, true, 1><<<dim3(G4H / 256, B_SZ / 32), 256, 0, stream>>>(
        xcat, Wg, bg, gates, B_SZ, G4H, KCAT, KCAT, KCAT, G4H);
    lstm_pointwise<<<(B_SZ * H_SZ) / 256, 256, 0, stream>>>(gates, hbuf, cbuf, hsb, t);
  }

  // 4) Output heads (O = hs @ W^T + b), written into d_out in return order
  float* out = (float*)d_out;
  float* out_char = out;
  float* out_bpe  = out + (size_t)MROWS * NC_SZ;
  float* out_wp   = out_bpe + (size_t)MROWS * BPE_SZ;

  wmma_gemm<false, true, true, 2><<<dim3((NC_SZ + 255) / 256, MROWS / 64), 256, 0, stream>>>(
      hsb, char_W, char_b, out_char, MROWS, NC_SZ, H_SZ, H_SZ, H_SZ, NC_SZ);
  wmma_gemm<false, true, true, 2><<<dim3((BPE_SZ + 255) / 256, MROWS / 64), 256, 0, stream>>>(
      hsb, bpe_W, bpe_b, out_bpe, MROWS, BPE_SZ, H_SZ, H_SZ, H_SZ, BPE_SZ);
  wmma_gemm<false, true, true, 2><<<dim3((WP_SZ + 255) / 256, MROWS / 64), 256, 0, stream>>>(
      hsb, wp_W, wp_b, out_wp, MROWS, WP_SZ, H_SZ, H_SZ, H_SZ, WP_SZ);
}
